// MultiHeadMaskedAttention_62706522521619
// MI455X (gfx1250) — matrix operand order repaired and checked by reading, not compiled, not run on hardware
//
#include <hip/hip_runtime.h>
#include <stdint.h>

// Problem constants (fixed by the reference).
#define DMODEL 512
#define NHEADS 8
#define DK 64
#define DV 64
#define BATCH 8
#define SEQ 1024

typedef __attribute__((ext_vector_type(16))) __bf16 v16bf;
typedef __attribute__((ext_vector_type(8)))  float  v8f;
typedef unsigned int u32x4 __attribute__((ext_vector_type(4)));
typedef int          i32x8 __attribute__((ext_vector_type(8)));
typedef int          i32x4 __attribute__((ext_vector_type(4)));

__device__ __forceinline__ v8f vzero8() {
  v8f z = {0.f, 0.f, 0.f, 0.f, 0.f, 0.f, 0.f, 0.f};
  return z;
}

__device__ __forceinline__ v8f wmma_bf16(v16bf a, v16bf b, v8f c) {
  // D = A(16x32 bf16) x B(32x16 bf16) + C(16x16 f32)
  return __builtin_amdgcn_wmma_f32_16x16x32_bf16(
      /*neg_a=*/false, a, /*neg_b=*/false, b,
      /*c_mod=*/(short)0, c, /*reuse_a=*/false, /*reuse_b=*/false);
}

// --- Tensor Data Mover descriptor builders (CDNA5 D#, CDNA5 ISA) -----
// Group 0: count=1 (valid user desc) | lds_addr | global_addr | type=2.
__device__ __forceinline__ u32x4 tdm_group0(unsigned lds_addr,
                                            unsigned long long gaddr) {
  u32x4 g;
  g[0] = 1u;                                     // count[1:0]=1, rest 0
  g[1] = lds_addr;                               // bits 63:32
  g[2] = (unsigned)(gaddr & 0xffffffffull);      // global_addr[31:0]
  g[3] = (unsigned)((gaddr >> 32) & 0x1ffffffull) | (2u << 30);  // addr[56:32] | type=2
  return g;
}
// Group 1: 2-D tile, data_size=2 bytes, no multicast/pad/iterate.
__device__ __forceinline__ i32x8 tdm_group1(unsigned tensor_d0, unsigned tensor_d1,
                                            unsigned tile_d0, unsigned tile_d1,
                                            unsigned long long d0_stride) {
  i32x8 g;
  g[0] = (int)(1u << 16);                                       // data_size=1 -> 2B
  g[1] = (int)((tensor_d0 & 0xffffu) << 16);                    // tensor_dim0[15:0]
  g[2] = (int)(((tensor_d0 >> 16) & 0xffffu) |
               ((tensor_d1 & 0xffffu) << 16));                  // d0 hi | d1 lo
  g[3] = (int)(((tensor_d1 >> 16) & 0xffffu) |
               ((tile_d0 & 0xffffu) << 16));                    // d1 hi | tile_dim0
  g[4] = (int)(tile_d1 & 0xffffu);                              // tile_dim1 (tile_dim2=0)
  g[5] = (int)(d0_stride & 0xffffffffull);                      // tensor_dim0_stride lo
  g[6] = (int)((d0_stride >> 32) & 0xffffull);                  // stride hi (dim1_stride=0)
  g[7] = 0;
  return g;
}

// ---------------------------------------------------------------------------
// Kernel 0: one-shot prep. Convert x -> bf16 and store weights bf16 TRANSPOSED
// so all GEMM B-fragments become contiguous per-lane loads.
//   xb    [B,N,512]          bf16
//   wqkvT [3][H][dk=64][512] bf16   (W^T per head)
//   woT   [dm=512][hv=512]   bf16   (Wo flat transposed)
// ---------------------------------------------------------------------------
__global__ __launch_bounds__(256)
void prep_kernel(const float* __restrict__ x,
                 const float* __restrict__ Wq,
                 const float* __restrict__ Wk,
                 const float* __restrict__ Wv,
                 const float* __restrict__ Wo,
                 __bf16* __restrict__ xb,
                 __bf16* __restrict__ wqkvT,
                 __bf16* __restrict__ woT) {
  const int idx = blockIdx.x * blockDim.x + threadIdx.x;
  if (idx < 3 * NHEADS * DK * DMODEL) {   // [m][h][dk][d] <- W[h][d][dk]
    const int d  = idx & (DMODEL - 1);
    const int dk = (idx >> 9) & (DK - 1);
    const int h  = (idx >> 15) & (NHEADS - 1);
    const int m  = idx >> 18;
    const float* W = (m == 0 ? Wq : (m == 1 ? Wk : Wv));
    wqkvT[idx] = (__bf16)W[((size_t)h * DMODEL + d) * DK + dk];
  }
  if (idx < DMODEL * DMODEL) {            // [dm][hv] <- Wo_flat[hv][dm]
    const int hv = idx & (DMODEL - 1);
    const int dm = idx >> 9;
    woT[idx] = (__bf16)Wo[(size_t)hv * DMODEL + dm];
  }
  if (idx < BATCH * SEQ * DMODEL) xb[idx] = (__bf16)x[idx];
}

// ---------------------------------------------------------------------------
// Kernel 1: Q/K/V projections.  One wave per (b, h, 16-row tile of N).
// A = bf16 x rows (contiguous), B = transposed bf16 weights (contiguous).
//   Q,K -> row-major bf16 [b,h,n,64];  V -> transposed bf16 [b,h,dv,n].
// ---------------------------------------------------------------------------
__global__ __launch_bounds__(256)
void qkv_proj_kernel(const __bf16* __restrict__ xb,
                     const __bf16* __restrict__ wqkvT,
                     __bf16* __restrict__ qb,
                     __bf16* __restrict__ kb,
                     __bf16* __restrict__ vtb) {
  const int wave = blockIdx.x * (blockDim.x >> 5) + (threadIdx.x >> 5);
  const int lane = threadIdx.x & 31;
  const int col  = lane & 15;
  const int hi   = lane >> 4;
  const int b  = wave >> 9;
  const int h  = (wave >> 6) & 7;
  const int n0 = (wave & 63) * 16;

  const __bf16* xrow = xb + ((size_t)b * SEQ + n0 + col) * DMODEL;
  const size_t bh = (size_t)b * NHEADS + h;

  #pragma unroll 1
  for (int m = 0; m < 3; ++m) {
    const __bf16* wt = wqkvT + (size_t)(m * NHEADS + h) * DK * DMODEL;
    v8f acc[4];
    acc[0] = vzero8(); acc[1] = vzero8(); acc[2] = vzero8(); acc[3] = vzero8();

    #pragma unroll 1
    for (int c = 0; c < DMODEL / 32; ++c) {
      v16bf a;
      const __bf16* xp = xrow + c * 32;
      #pragma unroll
      for (int i = 0; i < 8; ++i) {
        a[i]     = xp[8 * hi + i];
        a[8 + i] = xp[16 + 8 * hi + i];
      }
      #pragma unroll
      for (int t = 0; t < 4; ++t) {
        const __bf16* wp = wt + (size_t)(t * 16 + col) * DMODEL + c * 32;
        v16bf bf;
        #pragma unroll
        for (int i = 0; i < 8; ++i) {
          bf[i]     = wp[8 * hi + i];
          bf[8 + i] = wp[16 + 8 * hi + i];
        }
        acc[t] = wmma_bf16(a, bf, acc[t]);
      }
    }
    if (m < 2) {
      __bf16* dst = (m == 0 ? qb : kb) + bh * SEQ * DK;
      #pragma unroll
      for (int t = 0; t < 4; ++t)
        #pragma unroll
        for (int j = 0; j < 8; ++j)
          dst[(size_t)(n0 + 8 * hi + j) * DK + t * 16 + col] = (__bf16)acc[t][j];
    } else {
      __bf16* dst = vtb + bh * DV * SEQ;
      #pragma unroll
      for (int t = 0; t < 4; ++t)
        #pragma unroll
        for (int j = 0; j < 8; ++j)
          dst[(size_t)(t * 16 + col) * SEQ + n0 + 8 * hi + j] = (__bf16)acc[t][j];
    }
  }
}

// ---------------------------------------------------------------------------
// Kernel 2: fused masked flash attention with TDM staging.
// Block = 4 waves sharing one (b,h); each wave owns a 16-query tile.
// Per 32-key chunk: wave 0 DMAs K (32x64) and V^T (64x32) into LDS via
// tensor_load_to_lds, waits TENSORcnt, block barrier, then all waves run
// S^T = K·Q^T WMMAs (score C-fragments reused directly as the P A-fragment),
// shuffle-combined online softmax, and O += P·V WMMAs from LDS fragments.
// ---------------------------------------------------------------------------
__global__ __launch_bounds__(128)
void attn_kernel(const unsigned char* __restrict__ mask,  // [B,N,N], nonzero = masked
                 const __bf16* __restrict__ qb,
                 const __bf16* __restrict__ kb,
                 const __bf16* __restrict__ vtb,
                 __bf16* __restrict__ heads) {             // [B,N,H*DV] bf16
  extern __shared__ __bf16 smem[];          // [0..2047]=K 32x64, [2048..4095]=V^T 64x32
  __bf16* klds = smem;
  __bf16* vlds = smem + 2048;

  const int wid  = threadIdx.x >> 5;
  const int lane = threadIdx.x & 31;
  const int col  = lane & 15;
  const int hi   = lane >> 4;
  const int blk  = blockIdx.x;              // 1024 blocks = (b,h,qgroup of 4 tiles)
  const int b  = blk >> 7;
  const int h  = (blk >> 4) & 7;
  const int q0 = ((blk & 15) * 4 + wid) * 16;

  const size_t bh = (size_t)b * NHEADS + h;
  const __bf16* qbase = qb  + bh * SEQ * DK;
  const __bf16* kbp   = kb  + bh * SEQ * DK;
  const __bf16* vbase = vtb + bh * DV * SEQ;
  const unsigned char* mrow = mask + ((size_t)b * SEQ + q0 + col) * SEQ;

  // Q^T B-fragments (lane col = q-row, elems = d), contiguous loads.
  v16bf qf[2];
  #pragma unroll
  for (int c = 0; c < 2; ++c) {
    const __bf16* qp = qbase + (size_t)(q0 + col) * DK + c * 32;
    #pragma unroll
    for (int i = 0; i < 8; ++i) {
      qf[c][i]     = qp[8 * hi + i];
      qf[c][8 + i] = qp[16 + 8 * hi + i];
    }
  }

  float m_run = -__builtin_inff();
  float l_run = 0.0f;
  v8f o[4];
  o[0] = vzero8(); o[1] = vzero8(); o[2] = vzero8(); o[3] = vzero8();

  const i32x4 z4 = {0, 0, 0, 0};
  const i32x8 z8 = {0, 0, 0, 0, 0, 0, 0, 0};

  #pragma unroll 1
  for (int kc = 0; kc < SEQ / 32; ++kc) {
    const int kb0 = kc * 32;

    if (wid == 0) {
      // K chunk: 2-D tensor [SEQ x 64], tile 32 rows x 64 cols at row kb0.
      const unsigned long long kaddr =
          (unsigned long long)(uintptr_t)(kbp + (size_t)kb0 * DK);
      __builtin_amdgcn_tensor_load_to_lds(
          tdm_group0(0u, kaddr),
          tdm_group1(/*tensor_d0=*/DK, /*tensor_d1=*/SEQ,
                     /*tile_d0=*/DK, /*tile_d1=*/32, /*d0_stride=*/DK),
          z4, z4, z8, 0);
      // V^T chunk: 2-D tensor [64 x SEQ], tile 64 rows x 32 cols at col kb0.
      const unsigned long long vaddr =
          (unsigned long long)(uintptr_t)(vbase + kb0);
      __builtin_amdgcn_tensor_load_to_lds(
          tdm_group0(4096u, vaddr),
          tdm_group1(/*tensor_d0=*/SEQ, /*tensor_d1=*/DV,
                     /*tile_d0=*/32, /*tile_d1=*/DV, /*d0_stride=*/SEQ),
          z4, z4, z8, 0);
      __builtin_amdgcn_s_wait_tensorcnt(0);
    }
    __syncthreads();   // LDS tiles ready for all 4 waves

    // S^T tiles from LDS K + register Q^T  ->  C: col = q-row, regs = keys.
    v8f st[2];
    #pragma unroll
    for (int t2 = 0; t2 < 2; ++t2) {
      v8f s = vzero8();
      #pragma unroll
      for (int c = 0; c < 2; ++c) {
        const __bf16* kp = klds + (t2 * 16 + col) * DK + c * 32;
        v16bf ka;
        #pragma unroll
        for (int i = 0; i < 8; ++i) {
          ka[i]     = kp[8 * hi + i];
          ka[8 + i] = kp[16 + 8 * hi + i];
        }
        s = wmma_bf16(ka, qf[c], s);
      }
      const unsigned long long mb =
          *(const unsigned long long*)(mrow + kb0 + t2 * 16 + 8 * hi);
      #pragma unroll
      for (int j = 0; j < 8; ++j) {
        float sv = s[j] * 0.125f;                 // 1/sqrt(64)
        if ((mb >> (8 * j)) & 0xffull) sv = -1.0e9f;
        s[j] = sv;
      }
      st[t2] = s;
    }

    // Online softmax: lanes L and L+16 share a q-row with disjoint keys.
    float mloc = -__builtin_inff();
    #pragma unroll
    for (int t2 = 0; t2 < 2; ++t2)
      #pragma unroll
      for (int j = 0; j < 8; ++j) mloc = fmaxf(mloc, st[t2][j]);
    const float mpair = fmaxf(mloc, __shfl_xor(mloc, 16, 32));
    const float mnew  = fmaxf(m_run, mpair);
    const float scale = __expf(m_run - mnew);

    float lloc = 0.0f;
    #pragma unroll
    for (int t2 = 0; t2 < 2; ++t2)
      #pragma unroll
      for (int j = 0; j < 8; ++j) {
        const float p = __expf(st[t2][j] - mnew);
        st[t2][j] = p;
        lloc += p;
      }
    l_run = l_run * scale + (lloc + __shfl_xor(lloc, 16, 32));
    m_run = mnew;

    // P as A-fragment: elems 0..7 <- tile0 regs, 8..15 <- tile1 regs.
    v16bf pf;
    #pragma unroll
    for (int i = 0; i < 8; ++i) {
      pf[i]     = (__bf16)st[0][i];
      pf[8 + i] = (__bf16)st[1][i];
    }

    // Rescale O (reg j = q-row 8*hi+j; factor broadcast via ds_bpermute).
    #pragma unroll
    for (int j = 0; j < 8; ++j) {
      const float sj = __shfl(scale, 8 * hi + j, 32);
      #pragma unroll
      for (int t = 0; t < 4; ++t) o[t][j] *= sj;
    }

    // O += P · V using V^T fragments from LDS (contiguous per lane).
    #pragma unroll
    for (int t = 0; t < 4; ++t) {
      const __bf16* vp = vlds + (t * 16 + col) * 32;
      v16bf vf;
      #pragma unroll
      for (int i = 0; i < 8; ++i) {
        vf[i]     = vp[8 * hi + i];
        vf[8 + i] = vp[16 + 8 * hi + i];
      }
      o[t] = wmma_bf16(pf, vf, o[t]);
    }
    __syncthreads();   // readers done before next chunk's DMA overwrites LDS
  }

  // Normalize by l and store heads as [B, N, H*DV] (concat layout).
  #pragma unroll
  for (int j = 0; j < 8; ++j) {
    const float lj  = __shfl(l_run, 8 * hi + j, 32);
    const float inv = 1.0f / lj;
    const size_t row = (size_t)b * SEQ + q0 + 8 * hi + j;
    #pragma unroll
    for (int t = 0; t < 4; ++t)
      heads[row * (NHEADS * DV) + h * DV + t * 16 + col] = (__bf16)(o[t][j] * inv);
  }
}

// ---------------------------------------------------------------------------
// Kernel 3: output projection.  heads [8192,512] bf16 x woT -> out f32.
// One wave per 16x16 output tile; both fragments are contiguous bf16 loads.
// ---------------------------------------------------------------------------
__global__ __launch_bounds__(256)
void out_proj_kernel(const __bf16* __restrict__ woT,   // [dm][hv] bf16
                     const __bf16* __restrict__ heads,
                     float* __restrict__ out) {
  const int wave = blockIdx.x * (blockDim.x >> 5) + (threadIdx.x >> 5);
  const int lane = threadIdx.x & 31;
  const int col  = lane & 15;
  const int hi   = lane >> 4;
  const int rt = wave >> 5;
  const int ct = wave & 31;

  v8f acc = vzero8();
  #pragma unroll 1
  for (int c = 0; c < DMODEL / 32; ++c) {
    const __bf16* hp = heads + (size_t)(rt * 16 + col) * DMODEL + c * 32;
    v16bf a;
    #pragma unroll
    for (int i = 0; i < 8; ++i) {
      a[i]     = hp[8 * hi + i];
      a[8 + i] = hp[16 + 8 * hi + i];
    }
    const __bf16* wp = woT + (size_t)(ct * 16 + col) * DMODEL + c * 32;
    v16bf bf;
    #pragma unroll
    for (int i = 0; i < 8; ++i) {
      bf[i]     = wp[8 * hi + i];
      bf[8 + i] = wp[16 + 8 * hi + i];
    }
    acc = wmma_bf16(a, bf, acc);
  }
  #pragma unroll
  for (int j = 0; j < 8; ++j)
    out[(size_t)(rt * 16 + 8 * hi + j) * DMODEL + ct * 16 + col] = acc[j];
}

// ---------------------------------------------------------------------------
extern "C" void kernel_launch(void* const* d_in, const int* in_sizes, int n_in,
                              void* d_out, int out_size, void* d_ws, size_t ws_size,
                              hipStream_t stream) {
  const float*         x    = (const float*)d_in[0];
  const unsigned char* mask = (const unsigned char*)d_in[1];  // bool array, 1 B/elt
  const float*         Wq   = (const float*)d_in[2];
  const float*         Wk   = (const float*)d_in[3];
  const float*         Wv   = (const float*)d_in[4];
  const float*         Wo   = (const float*)d_in[5];
  float*               out  = (float*)d_out;

  // Workspace (bf16): q 8MB | k 8MB | vT 8MB | heads 8MB | xb 8MB | W^T ~2MB
  char* ws = (char*)d_ws;
  __bf16* qb    = (__bf16*)(ws);
  __bf16* kbuf  = (__bf16*)(ws + (8ull  << 20));
  __bf16* vtb   = (__bf16*)(ws + (16ull << 20));
  __bf16* hb    = (__bf16*)(ws + (24ull << 20));
  __bf16* xb    = (__bf16*)(ws + (32ull << 20));
  __bf16* wqkvT = (__bf16*)(ws + (40ull << 20));
  __bf16* woT   = (__bf16*)(ws + (42ull << 20));

  // Prep: covers max(B*N*DMODEL)=4.19M elements.
  prep_kernel<<<16384, 256, 0, stream>>>(x, Wq, Wk, Wv, Wo, xb, wqkvT, woT);
  // 4096 waves, 8 waves/block.
  qkv_proj_kernel<<<512, 256, 0, stream>>>(xb, wqkvT, qb, kbuf, vtb);
  // 1024 blocks x 4 waves sharing (b,h); 8KB dynamic LDS for K/V tiles.
  attn_kernel<<<1024, 128, 8192, stream>>>(mask, qb, kbuf, vtb, hb);
  // 16384 waves, 8 waves/block.
  out_proj_kernel<<<2048, 256, 0, stream>>>(woT, hb, out);
}
